// MLA_71975061946785
// MI455X (gfx1250) — matrix operand order repaired and checked by reading, not compiled, not run on hardware
//
#include <hip/hip_runtime.h>

// ---------------------------------------------------------------------------
// MLA forward on gfx1250 (MI455X): bf16 WMMA for all matmuls, fused causal
// flash attention. fp32 accumulate throughout.
// GEMM: one wave computes a 64x64 tile (4x4 WMMA tiles) for 4x data reuse.
// __launch_bounds__(32,1) => full VGPR budget, no accumulator spills.
// ---------------------------------------------------------------------------

#define BB 4
#define TT 2048
#define CC 1024
#define LL 256
#define HH 16
#define HSZ 64
#define MTOT (BB * TT)   // 8192 rows

typedef __attribute__((ext_vector_type(16))) __bf16 bf16x16;
typedef __attribute__((ext_vector_type(8)))  float  floatx8;

// ---- helpers --------------------------------------------------------------

static __device__ __forceinline__ __bf16 f2bf(float f) {
  unsigned u = __builtin_bit_cast(unsigned, f);
  unsigned r = u + 0x7FFFu + ((u >> 16) & 1u);     // round-to-nearest-even
  return __builtin_bit_cast(__bf16, (unsigned short)(r >> 16));
}

static __device__ __forceinline__ __bf16 bf_lo(unsigned d) {
  return __builtin_bit_cast(__bf16, (unsigned short)(d & 0xFFFFu));
}
static __device__ __forceinline__ __bf16 bf_hi(unsigned d) {
  return __builtin_bit_cast(__bf16, (unsigned short)(d >> 16));
}

// A fragment (16x32 bf16, K contiguous per row). row = this lane's row base.
// ISA layout: lanes 0-15 M=0..15; VGPR 0..3 -> K=8*half+2v, VGPR 4..7 -> K=16+8*half+2(v-4)
static __device__ __forceinline__ bf16x16 load_fragA_k32(const __bf16* row, int half) {
  bf16x16 f;
#pragma unroll
  for (int i = 0; i < 8; ++i) {
    int k = (i < 4) ? (8 * half + 2 * i) : (16 + 8 * half + 2 * (i - 4));
    unsigned d = *(const unsigned*)(row + k);
    f[2 * i]     = bf_lo(d);
    f[2 * i + 1] = bf_hi(d);
  }
  return f;
}

// B fragment (32x16 bf16) where element (k,n) is contiguous in k at col[k]
// (i.e. B supplied transposed / N-major). Same K order as A: VGPR 0..3 -> K=8*half+2v, VGPR 4..7 -> K=16+8*half+2(v-4).
static __device__ __forceinline__ bf16x16 load_fragB_k32(const __bf16* col, int half) {
  bf16x16 f;
#pragma unroll
  for (int i = 0; i < 8; ++i) {
    int k = (i < 4) ? (8 * half + 2 * i) : (16 + 8 * half + 2 * (i - 4));
    unsigned d = *(const unsigned*)(col + k);
    f[2 * i]     = bf_lo(d);
    f[2 * i + 1] = bf_hi(d);
  }
  return f;
}

// B fragment where element (k,n) sits at col[k*ld] (row-major K x N source).
static __device__ __forceinline__ bf16x16 load_fragB_strided(const __bf16* col, int ld, int half) {
  bf16x16 f;
#pragma unroll
  for (int i = 0; i < 8; ++i) {
    int k = (i < 4) ? (8 * half + 2 * i) : (16 + 8 * half + 2 * (i - 4));
    f[2 * i]     = col[(size_t)k * ld];
    f[2 * i + 1] = col[(size_t)(k + 1) * ld];
  }
  return f;
}

static __device__ __forceinline__ floatx8 wmma_bf16(bf16x16 a, bf16x16 b, floatx8 c) {
  return __builtin_amdgcn_wmma_f32_16x16x32_bf16(false, a, false, b, (short)0, c,
                                                 false, false);
}

// ---- conversion kernels ---------------------------------------------------

__global__ void k_f32_to_bf16(const float* __restrict__ in, __bf16* __restrict__ out, int n) {
  int i = blockIdx.x * blockDim.x + threadIdx.x;
  int stride = gridDim.x * blockDim.x;
  for (; i < n; i += stride) out[i] = f2bf(in[i]);
}

// in: [K,N] f32  ->  out: [N,K] bf16 (transposed)
__global__ void k_transpose_bf16(const float* __restrict__ in, __bf16* __restrict__ out,
                                 int K, int N) {
  int i = blockIdx.x * blockDim.x + threadIdx.x;
  int stride = gridDim.x * blockDim.x;
  int total = K * N;
  for (; i < total; i += stride) {
    int k = i / N, n = i - k * N;
    out[(size_t)n * K + k] = f2bf(in[i]);
  }
}

// ---- WMMA GEMM, 64x64 tile per wave ---------------------------------------
// C[M,N] = alpha * A[M,K] @ B[K,N], with B supplied transposed as Bt[N,K].
// One wave (32 threads) per 64x64 output tile: acc[4][4] 16x16 WMMA tiles.
// Per k-step: 4 A-fragments + 4 B-fragments feed 16 WMMAs (4x reuse each).
// grid = (N/64, M/64).
__global__ void __launch_bounds__(32, 1)
k_wmma_gemm64(const __bf16* __restrict__ A, const __bf16* __restrict__ Bt,
              void* __restrict__ Cout, int M, int N, int K,
              float alpha, int f32out) {
  const int n0 = blockIdx.x * 64;
  const int m0 = blockIdx.y * 64;
  const int l = threadIdx.x;
  const int half = l >> 4;
  const int lm = l & 15;

  const __bf16* arow[4];
  const __bf16* bcol[4];
#pragma unroll
  for (int mi = 0; mi < 4; ++mi) arow[mi] = A  + (size_t)(m0 + mi * 16 + lm) * K;
#pragma unroll
  for (int nj = 0; nj < 4; ++nj) bcol[nj] = Bt + (size_t)(n0 + nj * 16 + lm) * K;

  floatx8 acc[4][4] = {};

  for (int k0 = 0; k0 < K; k0 += 32) {
    bf16x16 af[4];
#pragma unroll
    for (int mi = 0; mi < 4; ++mi) af[mi] = load_fragA_k32(arow[mi] + k0, half);
#pragma unroll
    for (int nj = 0; nj < 4; ++nj) {
      bf16x16 bf = load_fragB_k32(bcol[nj] + k0, half);
#pragma unroll
      for (int mi = 0; mi < 4; ++mi) acc[mi][nj] = wmma_bf16(af[mi], bf, acc[mi][nj]);
    }
  }

#pragma unroll
  for (int mi = 0; mi < 4; ++mi)
#pragma unroll
    for (int nj = 0; nj < 4; ++nj)
#pragma unroll
      for (int r = 0; r < 8; ++r) {
        int m = m0 + mi * 16 + 8 * half + r;   // C/D layout: lanes0-15 M=r, lanes16-31 M=r+8
        int n = n0 + nj * 16 + lm;
        size_t idx = (size_t)m * N + n;
        float v = alpha * acc[mi][nj][r];
        if (f32out) ((float*)Cout)[idx] = v;
        else        ((__bf16*)Cout)[idx] = f2bf(v);
      }
}

// ---- fused causal flash attention ----------------------------------------
// q,k,v: [B*T, C] bf16, head h occupies cols [h*64, h*64+64). q pre-scaled by 1/8.
// One wave per (b, h, 16-query-row tile). grid = (T/16, H, B).
__global__ void __launch_bounds__(32, 1)
k_mla_attention(const __bf16* __restrict__ q, const __bf16* __restrict__ kk,
                const __bf16* __restrict__ vv, __bf16* __restrict__ y) {
  __shared__ __bf16 Plds[16 * 32];

  const int t0 = blockIdx.x * 16;
  const int h  = blockIdx.y;
  const int b  = blockIdx.z;
  const int l = threadIdx.x;
  const int half = l >> 4;
  const int lm = l & 15;

  const size_t rowbase = (size_t)b * TT;     // row index base
  const int hc = h * HSZ;                    // head column offset

  // Q fragments for rows t0..t0+15 (lane m = lm), d-chunks [0,32) and [32,64)
  const __bf16* qrow = q + (rowbase + t0 + lm) * CC + hc;
  bf16x16 qa0 = load_fragA_k32(qrow, half);
  bf16x16 qa1 = load_fragA_k32(qrow + 32, half);

  float mrow[8], lrow[8];
#pragma unroll
  for (int r = 0; r < 8; ++r) { mrow[r] = -__builtin_inff(); lrow[r] = 0.f; }
  floatx8 o[4] = {{}, {}, {}, {}};

  const int send = t0 + 15;
  for (int s0 = 0; s0 <= send; s0 += 32) {
    // ---- S = Q @ K^T for keys [s0, s0+32): two 16x16 tiles
    floatx8 s_acc[2];
#pragma unroll
    for (int j = 0; j < 2; ++j) {
      const __bf16* kcol = kk + (rowbase + s0 + j * 16 + lm) * CC + hc;
      floatx8 z = {};
      z = wmma_bf16(qa0, load_fragB_k32(kcol, half), z);
      z = wmma_bf16(qa1, load_fragB_k32(kcol + 32, half), z);
      s_acc[j] = z;
    }

    // ---- causal mask + online softmax (row m = 8*half + r, col n = lm)
    float pscale[8];
#pragma unroll
    for (int r = 0; r < 8; ++r) {
      const int qg = t0 + 8 * half + r;
      float s0v = s_acc[0][r];
      float s1v = s_acc[1][r];
      if (s0 + lm > qg)      s0v = -__builtin_inff();
      if (s0 + 16 + lm > qg) s1v = -__builtin_inff();

      float mx = fmaxf(s0v, s1v);
#pragma unroll
      for (int off = 1; off < 16; off <<= 1) mx = fmaxf(mx, __shfl_xor(mx, off, 32));
      float mnew = fmaxf(mrow[r], mx);
      float corr = __expf(mrow[r] - mnew);   // first block: exp(-inf)=0
      mrow[r] = mnew;
      pscale[r] = corr;

      float p0 = __expf(s0v - mnew);
      float p1 = __expf(s1v - mnew);
      float rs = p0 + p1;
#pragma unroll
      for (int off = 1; off < 16; off <<= 1) rs += __shfl_xor(rs, off, 32);
      lrow[r] = lrow[r] * corr + rs;
      s_acc[0][r] = p0;
      s_acc[1][r] = p1;
    }

    // rescale O accumulators by exp(m_old - m_new) per row
#pragma unroll
    for (int j = 0; j < 4; ++j)
#pragma unroll
      for (int r = 0; r < 8; ++r) o[j][r] *= pscale[r];

    // ---- P (16x32) -> LDS (row-major) to re-layout into an A fragment
#pragma unroll
    for (int j = 0; j < 2; ++j)
#pragma unroll
      for (int r = 0; r < 8; ++r)
        Plds[(8 * half + r) * 32 + j * 16 + lm] = f2bf(s_acc[j][r]);
    __syncthreads();   // single-wave block: safe, orders ds_store -> ds_load

    bf16x16 pa = load_fragA_k32(&Plds[lm * 32], half);

    // ---- O += P @ V   (V is [s,d] row-major: strided B fragment)
#pragma unroll
    for (int j = 0; j < 4; ++j) {
      const __bf16* vcol = vv + (rowbase + s0) * CC + hc + j * 16 + lm;
      o[j] = wmma_bf16(pa, load_fragB_strided(vcol, CC, half), o[j]);
    }
    __syncthreads();   // protect Plds before next iteration overwrites
  }

  // ---- epilogue: normalize and store y[b, t, h*64 + d]
  float inv[8];
#pragma unroll
  for (int r = 0; r < 8; ++r) inv[r] = 1.f / lrow[r];
#pragma unroll
  for (int j = 0; j < 4; ++j)
#pragma unroll
    for (int r = 0; r < 8; ++r) {
      size_t idx = (rowbase + t0 + 8 * half + r) * CC + hc + j * 16 + lm;
      y[idx] = f2bf(o[j][r] * inv[r]);
    }
}

// ---------------------------------------------------------------------------

extern "C" void kernel_launch(void* const* d_in, const int* in_sizes, int n_in,
                              void* d_out, int out_size, void* d_ws, size_t ws_size,
                              hipStream_t stream) {
  const float* x       = (const float*)d_in[0];   // [B,T,C]
  const float* Wq      = (const float*)d_in[1];   // [C,C]
  const float* Wkvd    = (const float*)d_in[2];   // [C,L]
  const float* Wkup    = (const float*)d_in[3];   // [L,C]
  const float* Wvup    = (const float*)d_in[4];   // [L,C]
  const float* Wout    = (const float*)d_in[5];   // [C,C]
  float* out = (float*)d_out;

  // workspace layout (bf16 elements)
  __bf16* ws = (__bf16*)d_ws;
  size_t off = 0;
  __bf16* xb     = ws + off; off += (size_t)MTOT * CC;   // x bf16
  __bf16* qb     = ws + off; off += (size_t)MTOT * CC;   // q (pre-scaled)
  __bf16* kb     = ws + off; off += (size_t)MTOT * CC;   // k
  __bf16* vb     = ws + off; off += (size_t)MTOT * CC;   // v
  __bf16* yb     = ws + off; off += (size_t)MTOT * CC;   // attention out
  __bf16* latb   = ws + off; off += (size_t)MTOT * LL;   // latent kv
  __bf16* WqT    = ws + off; off += (size_t)CC * CC;
  __bf16* WkvdT  = ws + off; off += (size_t)CC * LL;
  __bf16* WkupT  = ws + off; off += (size_t)LL * CC;
  __bf16* WvupT  = ws + off; off += (size_t)LL * CC;
  __bf16* WoutT  = ws + off; off += (size_t)CC * CC;

  // 1) precision conversion (+ weight transpose so WMMA B-fragments are contiguous)
  k_f32_to_bf16<<<2048, 256, 0, stream>>>(x, xb, MTOT * CC);
  k_transpose_bf16<<<1024, 256, 0, stream>>>(Wq,   WqT,   CC, CC);
  k_transpose_bf16<<<512,  256, 0, stream>>>(Wkvd, WkvdT, CC, LL);
  k_transpose_bf16<<<512,  256, 0, stream>>>(Wkup, WkupT, LL, CC);
  k_transpose_bf16<<<512,  256, 0, stream>>>(Wvup, WvupT, LL, CC);
  k_transpose_bf16<<<1024, 256, 0, stream>>>(Wout, WoutT, CC, CC);

  // 2) projections (q folded with 1/sqrt(HS) = 0.125)
  k_wmma_gemm64<<<dim3(CC / 64, MTOT / 64), 32, 0, stream>>>(xb, WqT, qb,
      MTOT, CC, CC, 0.125f, 0);
  k_wmma_gemm64<<<dim3(LL / 64, MTOT / 64), 32, 0, stream>>>(xb, WkvdT, latb,
      MTOT, LL, CC, 1.0f, 0);
  k_wmma_gemm64<<<dim3(CC / 64, MTOT / 64), 32, 0, stream>>>(latb, WkupT, kb,
      MTOT, CC, LL, 1.0f, 0);
  k_wmma_gemm64<<<dim3(CC / 64, MTOT / 64), 32, 0, stream>>>(latb, WvupT, vb,
      MTOT, CC, LL, 1.0f, 0);

  // 3) fused causal attention
  k_mla_attention<<<dim3(TT / 16, HH, BB), 32, 0, stream>>>(qb, kb, vb, yb);

  // 4) output projection (fp32 result)
  k_wmma_gemm64<<<dim3(CC / 64, MTOT / 64), 32, 0, stream>>>(yb, WoutT, out,
      MTOT, CC, CC, 1.0f, 1);
}
